// SelfAttention_65859028517345
// MI455X (gfx1250) — matrix operand order repaired and checked by reading, not compiled, not run on hardware
//
#include <hip/hip_runtime.h>

#define HID 768
#define ADIM 64
#define SEQ 2048
#define NB 8
#define NROWS (NB * SEQ)      // 16384
#define NTILES (NROWS / 16)   // 1024

typedef __attribute__((ext_vector_type(16))) __bf16 v16bf;
typedef __attribute__((ext_vector_type(2)))  __bf16 bf2;
typedef __attribute__((ext_vector_type(8)))  float  v8f;
typedef __attribute__((ext_vector_type(4)))  unsigned int v4u;

union BF16x16 {
    v16bf v;
    unsigned int u[8];
};

// Native bf16 conversion (RTNE) — lowers to v_cvt_pk_bf16_f32.
__device__ __forceinline__ unsigned pkbf(float a, float b) {
    bf2 t;
    t.x = (__bf16)a;
    t.y = (__bf16)b;
    return __builtin_bit_cast(unsigned, t);
}

__device__ __forceinline__ v8f wmma_bf16(const BF16x16& a, const BF16x16& b, v8f c) {
    return __builtin_amdgcn_wmma_f32_16x16x32_bf16(
        false, a.v, false, b.v, (short)0, c, false, false);
}

__device__ __forceinline__ void load_dw4(const unsigned short* p, unsigned int* dst) {
    const unsigned int* pi = (const unsigned int*)p;
#pragma unroll
    for (int j = 0; j < 4; ++j) dst[j] = pi[j];
}
__device__ __forceinline__ void load_dw8(const unsigned short* p, unsigned int* dst) {
    const unsigned int* pi = (const unsigned int*)p;
#pragma unroll
    for (int j = 0; j < 8; ++j) dst[j] = pi[j];
}

// ---------------------------------------------------------------------------
// Kernel 0: convert Wq|Wk|Wv (each [64,768] fp32) to contiguous bf16 buffer.
// ---------------------------------------------------------------------------
__global__ __launch_bounds__(256) void cvt_w_kernel(
    const float* __restrict__ wq, const float* __restrict__ wk,
    const float* __restrict__ wv, unsigned int* __restrict__ Wbf) {
    int i = blockIdx.x * 256 + threadIdx.x;     // pair index
    const int perPairs = (ADIM * HID) / 2;      // 24576 pairs per matrix
    if (i >= 3 * perPairs) return;
    int p = i / perPairs, j = (i - p * perPairs) * 2;
    const float* src = (p == 0) ? wq : (p == 1) ? wk : wv;
    Wbf[i] = pkbf(src[j], src[j + 1]);
}

// ---------------------------------------------------------------------------
// Kernel 1: projections. One wave computes a 16-row tile x 64 dims for
// q, k, v.  Q (pre-scaled by 1/sqrt(64)*log2e), K row-major bf16
// [B*S][64]; V transposed bf16 [B][64][S] for contiguous PV B-loads.
// ---------------------------------------------------------------------------
__global__ __launch_bounds__(256) void proj_kernel(
    const float* __restrict__ q_in, const float* __restrict__ k_in,
    const float* __restrict__ v_in, const unsigned short* __restrict__ Wbf,
    unsigned short* __restrict__ Qbf, unsigned short* __restrict__ Kbf,
    unsigned short* __restrict__ Vt) {
    const int wave = threadIdx.x >> 5;
    const int lane = threadIdx.x & 31;
    const int tile = blockIdx.x * 8 + wave;          // 0..1023
    const int m    = lane & 15;
    const int half = lane >> 4;
    const int kA   = half * 8;                       // A-layout K base
    const long rowBase = (long)tile * 16;
    const int  b    = tile >> 7;                     // 128 tiles per batch
    const int  qInB = (tile & 127) * 16;

    const float* xs[3] = {q_in, k_in, v_in};
    // exp2-domain softmax: fold scale*log2(e) into Q.
    const float QSCALE = 0.125f * 1.44269504088896340736f;

    for (int p = 0; p < 3; ++p) {
        const float* xrow = xs[p] + (rowBase + m) * HID;
        const unsigned short* W = Wbf + (size_t)p * (ADIM * HID);

        v8f acc[4];
#pragma unroll
        for (int nt = 0; nt < 4; ++nt) acc[nt] = (v8f){};

        for (int k0 = 0; k0 < HID; k0 += 32) {
            // A tile: rows of x, fp32 -> bf16 (packed cvt), contiguous k = 16*half + s order.
            BF16x16 A;
#pragma unroll
            for (int j = 0; j < 4; ++j) {
                int k1 = k0 + 2 * kA + 2 * j;
                int k2 = k0 + 8 + 2 * kA + 2 * j;
                A.u[j]     = pkbf(xrow[k1], xrow[k1 + 1]);
                A.u[4 + j] = pkbf(xrow[k2], xrow[k2 + 1]);
            }
#pragma unroll
            for (int nt = 0; nt < 4; ++nt) {
                // B tile: B[k][n] = W[n][k]; lane holds column n = nt*16+m,
                // K half split by lane half -> contiguous 16 bf16.
                BF16x16 Bm;
                load_dw8(W + (size_t)(nt * 16 + m) * HID + k0 + half * 16, Bm.u);
                acc[nt] = wmma_bf16(A, Bm, acc[nt]);
            }
        }

        if (p < 2) {  // Q, K: row-major [B*S][64] bf16
            __bf16* outp = (__bf16*)((p == 0) ? Qbf : Kbf);
            const float sc = (p == 0) ? QSCALE : 1.0f;
#pragma unroll
            for (int nt = 0; nt < 4; ++nt)
#pragma unroll
                for (int r = 0; r < 8; ++r) {
                    long row = rowBase + half * 8 + r;   // C-layout M
                    outp[row * ADIM + nt * 16 + m] = (__bf16)(acc[nt][r] * sc);
                }
        } else {      // V transposed: [B][64][S] bf16, packed 16B stores
#pragma unroll
            for (int nt = 0; nt < 4; ++nt) {
                int n = nt * 16 + m;
                v4u pk;
#pragma unroll
                for (int r = 0; r < 4; ++r)
                    pk[r] = pkbf(acc[nt][2 * r], acc[nt][2 * r + 1]);
                unsigned short* vp =
                    Vt + ((long)b * ADIM + n) * SEQ + qInB + half * 8;
                *reinterpret_cast<v4u*>(vp) = pk;
            }
        }
    }
}

// ---------------------------------------------------------------------------
// Kernel 2: fused flash attention. One wave per 16-query tile; online
// softmax (exp2 domain) over kv in steps of 32.
//  - Score tiles split even/odd kv rows so each lane owns adjacent P
//    columns (2m, 2m+1) -> packed b32 LDS stores.
//  - Softmax denominator accumulated by an extra WMMA against an
//    all-ones B matrix (same rescale recurrence as the output tiles).
// ---------------------------------------------------------------------------
__global__ __launch_bounds__(256) void attn_kernel(
    const unsigned short* __restrict__ Qbf,
    const unsigned short* __restrict__ Kbf,
    const unsigned short* __restrict__ Vt,
    float* __restrict__ out) {
    __shared__ unsigned short lds[8 * 16 * 32];      // 8 waves x 1KB
    const int wave = threadIdx.x >> 5;
    const int lane = threadIdx.x & 31;
    unsigned short* P = lds + wave * 512;
    const int tile = blockIdx.x * 8 + wave;          // 0..1023
    const int b    = tile >> 7;
    const int qInB = (tile & 127) * 16;
    const int m    = lane & 15;
    const int half = lane >> 4;
    const int kA   = half * 8;

    // Q held in registers as two 16x32 A tiles (head dims 0..31, 32..63).
    BF16x16 Aq0, Aq1;
    {
        const unsigned short* qrow =
            Qbf + ((long)b * SEQ + qInB + m) * ADIM;
        load_dw4(qrow + 2 * kA, Aq0.u);
        load_dw4(qrow + 8 + 2 * kA, Aq0.u + 4);
        load_dw4(qrow + 32 + 2 * kA, Aq1.u);
        load_dw4(qrow + 40 + 2 * kA, Aq1.u + 4);
    }

    // All-ones bf16 B matrix for row-sum accumulation.
    BF16x16 Bones;
#pragma unroll
    for (int j = 0; j < 8; ++j) Bones.u[j] = 0x3F803F80u;   // (1.0, 1.0)

    float rmax[8];
#pragma unroll
    for (int r = 0; r < 8; ++r) rmax[r] = -1e30f;
    v8f acc[5];                                      // [0..3] out, [4] rowsum
#pragma unroll
    for (int nt = 0; nt < 5; ++nt) acc[nt] = (v8f){};

    const unsigned short* kbase = Kbf + (long)b * SEQ * ADIM;

    for (int kv0 = 0; kv0 < SEQ; kv0 += 32) {
        // Prefetch next iteration's K rows (gfx1250 global_prefetch path).
        if (kv0 + 32 < SEQ)
            __builtin_prefetch(kbase + (size_t)(kv0 + 32 + lane) * ADIM, 0, 3);

        // ---- scores: tile s0 = even kv rows (kv0+2n), s1 = odd rows ----
        v8f s0 = (v8f){}, s1 = (v8f){};
        {
            BF16x16 B0, B1;
            const unsigned short* kre =
                kbase + (size_t)(kv0 + 2 * m) * ADIM + half * 16;
            load_dw8(kre, B0.u);        // head dims 0..31 (half-split)
            load_dw8(kre + 32, B1.u);   // head dims 32..63
            s0 = wmma_bf16(Aq0, B0, s0);
            s0 = wmma_bf16(Aq1, B1, s0);
            const unsigned short* kro = kre + ADIM;   // odd row
            load_dw8(kro, B0.u);
            load_dw8(kro + 32, B1.u);
            s1 = wmma_bf16(Aq0, B0, s1);
            s1 = wmma_bf16(Aq1, B1, s1);
        }

        // ---- online softmax in exp2 domain (scale folded into Q) ----
        // Rows live per C-layout: M = half*8 + r; row's 16 cols = 16 lanes
        // of this half, so xor-masks 1,2,4,8 reduce without mixing halves.
        unsigned int* P32 = (unsigned int*)P;
#pragma unroll
        for (int r = 0; r < 8; ++r) {
            float a = s0[r];            // col 2m
            float c = s1[r];            // col 2m+1
            float mx = fmaxf(a, c);
#pragma unroll
            for (int off = 1; off < 16; off <<= 1)
                mx = fmaxf(mx, __shfl_xor(mx, off, 32));
            float mnew = fmaxf(rmax[r], mx);
            float corr = exp2f(rmax[r] - mnew);
            rmax[r] = mnew;
            float p0 = exp2f(a - mnew);
            float p1 = exp2f(c - mnew);
            int Mrow = half * 8 + r;
            P32[Mrow * 16 + m] = pkbf(p0, p1);       // cols 2m, 2m+1
#pragma unroll
            for (int nt = 0; nt < 5; ++nt) acc[nt][r] *= corr;
        }
        asm volatile("" ::: "memory");   // keep DS store->load order

        // ---- reload P in A layout; PV + rowsum accumulation ----
        BF16x16 Ap;
        const unsigned short* pr = P + m * 32 + 2 * kA;
        load_dw4(pr, Ap.u);
        load_dw4(pr + 8, Ap.u + 4);
#pragma unroll
        for (int nt = 0; nt < 4; ++nt) {
            BF16x16 Bv;   // B[k][n] = Vt[b][n][kv0+k], contiguous in k
            load_dw8(Vt + ((long)b * ADIM + nt * 16 + m) * SEQ + kv0 + half * 16,
                     Bv.u);
            acc[nt] = wmma_bf16(Ap, Bv, acc[nt]);
        }
        acc[4] = wmma_bf16(Ap, Bones, acc[4]);       // rowsum(P) per row
    }

    // ---- epilogue: normalize and store fp32 [B,S,64] ----
#pragma unroll
    for (int r = 0; r < 8; ++r) {
        float inv = 1.0f / acc[4][r];
        long row = (long)b * SEQ + qInB + half * 8 + r;
#pragma unroll
        for (int nt = 0; nt < 4; ++nt)
            out[row * ADIM + nt * 16 + m] = acc[nt][r] * inv;
    }
}

// ---------------------------------------------------------------------------
extern "C" void kernel_launch(void* const* d_in, const int* in_sizes, int n_in,
                              void* d_out, int out_size, void* d_ws, size_t ws_size,
                              hipStream_t stream) {
    (void)in_sizes; (void)n_in; (void)out_size; (void)ws_size;
    const float* q  = (const float*)d_in[0];
    const float* k  = (const float*)d_in[1];
    const float* v  = (const float*)d_in[2];
    const float* wq = (const float*)d_in[3];
    const float* wk = (const float*)d_in[4];
    const float* wv = (const float*)d_in[5];

    unsigned char* ws = (unsigned char*)d_ws;
    unsigned short* Wbf = (unsigned short*)ws;                           // 288 KB
    unsigned short* Qbf = (unsigned short*)(ws + (size_t)(512 << 10));   // 2 MB
    unsigned short* Kbf = (unsigned short*)(ws + (size_t)(512 << 10) + (size_t)(2 << 20));
    unsigned short* Vt  = (unsigned short*)(ws + (size_t)(512 << 10) + (size_t)(4 << 20));
    float* out = (float*)d_out;

    const int wpairs = (3 * ADIM * HID) / 2;
    cvt_w_kernel<<<(wpairs + 255) / 256, 256, 0, stream>>>(
        wq, wk, wv, (unsigned int*)Wbf);
    proj_kernel<<<NTILES / 8, 256, 0, stream>>>(q, k, v, Wbf, Qbf, Kbf, Vt);
    attn_kernel<<<NTILES / 8, 256, 0, stream>>>(Qbf, Kbf, Vt, out);
}
